// TransformerBlock_84473416778117
// MI455X (gfx1250) — matrix operand order repaired and checked by reading, not compiled, not run on hardware
//
#include <hip/hip_runtime.h>
#include <hip/hip_bf16.h>
#include <math.h>

typedef __bf16 bf16;
typedef bf16  v16bf __attribute__((ext_vector_type(16)));
typedef float v8f   __attribute__((ext_vector_type(8)));
typedef unsigned short u16;
typedef u16  v8u16  __attribute__((ext_vector_type(8)));
typedef int  v4i    __attribute__((vector_size(16)));   // matches builtin param type

union Frag16 { v16bf bf; v8u16 u[2]; };

__device__ __forceinline__ u16 f2b(float f) {
    unsigned int x = __float_as_uint(f);
    unsigned int r = x + 0x7FFFu + ((x >> 16) & 1u);   // round-to-nearest-even
    return (u16)(r >> 16);
}

// ---- async global->LDS copy (CDNA5 GLOBAL_LOAD_ASYNC_TO_LDS_B128), with fallback
#if defined(__has_builtin)
#if __has_builtin(__builtin_amdgcn_global_load_async_to_lds_b128) && \
    __has_builtin(__builtin_amdgcn_s_wait_asynccnt)
#define USE_ASYNC_LDS 1
#endif
#endif

__device__ __forceinline__ void cp16_g2l(const u16* g, u16* l) {
#ifdef USE_ASYNC_LDS
    __builtin_amdgcn_global_load_async_to_lds_b128(
        (__attribute__((address_space(1))) v4i*)g,
        (__attribute__((address_space(3))) v4i*)l, 0, 0);
#else
    *(v8u16*)l = *(const v8u16*)g;
#endif
}
__device__ __forceinline__ void wait_cp() {
#ifdef USE_ASYNC_LDS
    __builtin_amdgcn_s_wait_asynccnt(0);
#endif
}

// ---------------------------------------------------------------- convert + transpose
// in: f32 [K][N] row-major  ->  out: bf16 [N][K] row-major
__global__ __launch_bounds__(256) void k_cvt_transpose(
    const float* __restrict__ in, u16* __restrict__ out, int K, int N)
{
    __shared__ float tile[32][33];
    int tx = threadIdx.x, ty = threadIdx.y;      // (32,8)
    int k0 = blockIdx.y * 32, n0 = blockIdx.x * 32;
    #pragma unroll
    for (int i = 0; i < 4; i++)
        tile[ty * 4 + i][tx] = in[(size_t)(k0 + ty * 4 + i) * N + n0 + tx];
    __syncthreads();
    #pragma unroll
    for (int i = 0; i < 4; i++) {
        int n = ty * 4 + i;
        out[(size_t)(n0 + n) * K + k0 + tx] = f2b(tile[tx][n]);
    }
}

// ---------------------------------------------------------------- layernorm -> bf16
__global__ __launch_bounds__(256) void k_layernorm_bf16(
    const float* __restrict__ x, const float* __restrict__ scale,
    const float* __restrict__ shift, u16* __restrict__ out, int C)
{
    __shared__ float red[256];
    int row = blockIdx.x;
    const float* xr = x + (size_t)row * C;
    float s = 0.f;
    for (int c = threadIdx.x; c < C; c += 256) s += xr[c];
    red[threadIdx.x] = s; __syncthreads();
    for (int off = 128; off > 0; off >>= 1) {
        if (threadIdx.x < off) red[threadIdx.x] += red[threadIdx.x + off];
        __syncthreads();
    }
    float mean = red[0] / (float)C;
    __syncthreads();
    float v = 0.f;
    for (int c = threadIdx.x; c < C; c += 256) { float d = xr[c] - mean; v += d * d; }
    red[threadIdx.x] = v; __syncthreads();
    for (int off = 128; off > 0; off >>= 1) {
        if (threadIdx.x < off) red[threadIdx.x] += red[threadIdx.x + off];
        __syncthreads();
    }
    float rstd = rsqrtf(red[0] / (float)C + 1e-5f);
    u16* orow = out + (size_t)row * C;
    for (int c = threadIdx.x; c < C; c += 256)
        orow[c] = f2b((xr[c] - mean) * rstd * scale[c] + shift[c]);
}

// ---------------------------------------------------------------- tiled WMMA GEMM
// C[M,N] = A[M,K](bf16 row-major) x B[K,N], with B supplied TRANSPOSED: BT[N][K].
// EPI 0: store bf16
//     1: +bias, +residual(f32), store f32
//     2: +bias, gelu(tanh), store bf16
//     3: store bf16 transposed per-head: out[(b*1024+gn)*2048 + seq]  (V layout)
#define TM 128
#define TN 128
#define TK 32
#define LST 40   // LDS row stride in u16 (80 B: 16B-aligned chunks, skewed banks)

template<int EPI>
__global__ __launch_bounds__(256) void k_gemm_bf16(
    const u16* __restrict__ A, const u16* __restrict__ BT, int M, int N, int K,
    const float* __restrict__ bias, const float* __restrict__ resid,
    u16* __restrict__ outb, float* __restrict__ outf)
{
    __shared__ u16 As[2][TM * LST];
    __shared__ u16 Bs[2][TN * LST];

    int t = threadIdx.x;
    int lane = t & 31, wave = t >> 5;
    int hi = lane >> 4, lo = lane & 15;
    int wm = (wave & 3) * 32;         // 4 waves along M
    int wn = (wave >> 2) * 64;        // 2 waves along N
    int bm0 = blockIdx.x * TM;
    int bn0 = blockIdx.y * TN;

    v8f zero = {0.f,0.f,0.f,0.f,0.f,0.f,0.f,0.f};
    v8f acc[2][4];
    for (int i = 0; i < 2; i++) for (int j = 0; j < 4; j++) acc[i][j] = zero;

    auto stage = [&](int buf, int k0) {
        #pragma unroll
        for (int i = 0; i < 2; i++) {
            int c = t + 256 * i;
            int row = c >> 2;             // 0..127
            int col = (c & 3) * 8;        // 0,8,16,24
            cp16_g2l(&A [(size_t)(bm0 + row) * K + k0 + col], &As[buf][row * LST + col]);
            cp16_g2l(&BT[(size_t)(bn0 + row) * K + k0 + col], &Bs[buf][row * LST + col]);
        }
    };

    stage(0, 0);
    wait_cp();
    __syncthreads();

    int nk = K / TK;
    for (int idx = 0; idx < nk; idx++) {
        int buf = idx & 1;
        bool more = (idx + 1 < nk);
        if (more) stage(1 - buf, (idx + 1) * TK);   // overlap with WMMA below

        Frag16 af[2], bfr[4];
        #pragma unroll
        for (int mi = 0; mi < 2; mi++) {
            int r = wm + mi * 16 + lo;
            af[mi].u[0] = *(const v8u16*)&As[buf][r * LST + hi * 8];
            af[mi].u[1] = *(const v8u16*)&As[buf][r * LST + hi * 8 + 16];
        }
        #pragma unroll
        for (int ni = 0; ni < 4; ni++) {
            int n = wn + ni * 16 + lo;
            bfr[ni].u[0] = *(const v8u16*)&Bs[buf][n * LST + hi * 8];
            bfr[ni].u[1] = *(const v8u16*)&Bs[buf][n * LST + hi * 8 + 16];
        }
        #pragma unroll
        for (int mi = 0; mi < 2; mi++)
            #pragma unroll
            for (int ni = 0; ni < 4; ni++)
                acc[mi][ni] = __builtin_amdgcn_wmma_f32_16x16x32_bf16(
                    false, af[mi].bf, false, bfr[ni].bf,
                    (short)0, acc[mi][ni], false, false);

        if (more) {
            wait_cp();
            __syncthreads();
        }
    }

    // epilogue: C fragment layout: VGPR r -> M = r + 8*hi, N = lo
    #pragma unroll
    for (int mi = 0; mi < 2; mi++) {
        #pragma unroll
        for (int ni = 0; ni < 4; ni++) {
            int gn = bn0 + wn + ni * 16 + lo;
            #pragma unroll
            for (int r = 0; r < 8; r++) {
                int gm = bm0 + wm + mi * 16 + r + 8 * hi;
                float vv = acc[mi][ni][r];
                if (EPI == 1 || EPI == 2) vv += bias[gn];
                if (EPI == 2) {
                    float u = 0.7978845608028654f * (vv + 0.044715f * vv * vv * vv);
                    vv = 0.5f * vv * (1.f + tanhf(u));
                }
                if (EPI == 1) {
                    vv += resid[(size_t)gm * N + gn];
                    outf[(size_t)gm * N + gn] = vv;
                } else if (EPI == 3) {
                    int bb  = gm >> 11;         // batch (SEQ=2048)
                    int seq = gm & 2047;
                    outb[((size_t)bb * 1024 + gn) * 2048 + seq] = f2b(vv);
                } else {
                    outb[(size_t)gm * N + gn] = f2b(vv);
                }
            }
        }
    }
}

// ---------------------------------------------------------------- flash attention
// one wave per (batch, head, 16-query tile); key tiles of 32.
// q,k: bf16 [B*SEQ][1024] row-major; vt: bf16 [B][H][64][SEQ] (per-head transposed)
#define SEQ 2048
#define HD  64
#define DM  1024

__global__ __launch_bounds__(32) void k_attention(
    const u16* __restrict__ q, const u16* __restrict__ k, const u16* __restrict__ vt,
    u16* __restrict__ ctx)
{
    __shared__ float Sb[16 * 33];     // S tile 16q x 32keys
    __shared__ u16   Pb[16 * LST];    // P tile bf16
    __shared__ float fac[16];

    int lane = threadIdx.x;
    int hi = lane >> 4, lo = lane & 15;
    int bid = blockIdx.x;             // b*H*128 + h*128 + qt
    int qt = bid & 127;
    int hh = (bid >> 7) & 15;
    int b  = bid >> 11;
    int q0 = qt * 16;
    size_t rowbase = (size_t)b * SEQ;
    int colbase = hh * HD;
    size_t vbase = ((size_t)b * 1024 + colbase) * SEQ;

    // Q A-fragments, two K-chunks of 32 over Dh=64, loaded straight from global
    Frag16 qf[2];
    {
        size_t qrow = (rowbase + q0 + lo) * DM + colbase;
        #pragma unroll
        for (int kc = 0; kc < 2; kc++) {
            qf[kc].u[0] = *(const v8u16*)&q[qrow + kc * 32 + hi * 8];
            qf[kc].u[1] = *(const v8u16*)&q[qrow + kc * 32 + hi * 8 + 16];
        }
    }

    v8f zero = {0.f,0.f,0.f,0.f,0.f,0.f,0.f,0.f};
    v8f cacc[4];
    for (int i = 0; i < 4; i++) cacc[i] = zero;
    float mrun = -1e30f, lrun = 0.f;  // lanes 0..15 own row = lane

    int kt_end = q0 + 16;
    for (int kt0 = 0; kt0 < kt_end; kt0 += 32) {
        // ---- load ALL K fragments first (8 x b128 in flight), then WMMA chain
        Frag16 kf[2][2];
        #pragma unroll
        for (int ni = 0; ni < 2; ni++) {
            #pragma unroll
            for (int kc = 0; kc < 2; kc++) {
                size_t kr = (rowbase + kt0 + ni * 16 + lo) * DM + colbase + kc * 32 + hi * 8;
                kf[ni][kc].u[0] = *(const v8u16*)&k[kr];
                kf[ni][kc].u[1] = *(const v8u16*)&k[kr + 16];
            }
        }
        v8f sA[2] = {zero, zero};
        #pragma unroll
        for (int ni = 0; ni < 2; ni++)
            #pragma unroll
            for (int kc = 0; kc < 2; kc++)
                sA[ni] = __builtin_amdgcn_wmma_f32_16x16x32_bf16(
                    false, qf[kc].bf, false, kf[ni][kc].bf, (short)0, sA[ni], false, false);

        // scale + causal mask -> LDS
        #pragma unroll
        for (int ni = 0; ni < 2; ni++) {
            int key = kt0 + ni * 16 + lo;
            #pragma unroll
            for (int r = 0; r < 8; r++) {
                int m = r + 8 * hi;
                float val = sA[ni][r] * 0.125f;          // 1/sqrt(64)
                if (key > q0 + m) val = -1e30f;
                Sb[m * 33 + ni * 16 + lo] = val;
            }
        }
        // prefetch all V fragments while softmax round-trips through LDS
        Frag16 vf[4];
        #pragma unroll
        for (int ni = 0; ni < 4; ni++) {
            size_t vr = vbase + (size_t)(ni * 16 + lo) * SEQ + kt0 + hi * 16;
            vf[ni].u[0] = *(const v8u16*)&vt[vr];
            vf[ni].u[1] = *(const v8u16*)&vt[vr + 8];
        }
        __syncthreads();
        // online softmax, one row per lane 0..15
        if (lane < 16) {
            int m = lane;
            float rowmax = -1e30f;
            for (int j = 0; j < 32; j++) rowmax = fmaxf(rowmax, Sb[m * 33 + j]);
            float newm = fmaxf(mrun, rowmax);
            float corr = __expf(mrun - newm);
            float sum = 0.f;
            for (int j = 0; j < 32; j++) {
                float p = __expf(Sb[m * 33 + j] - newm);
                sum += p;
                Pb[m * LST + j] = f2b(p);
            }
            lrun = lrun * corr + sum;
            mrun = newm;
            fac[m] = corr;
        }
        __syncthreads();
        // rescale accumulators by row correction factor
        #pragma unroll
        for (int r = 0; r < 8; r++) {
            float f = fac[r + 8 * hi];
            #pragma unroll
            for (int ni = 0; ni < 4; ni++) cacc[ni][r] *= f;
        }
        // P A-fragment from LDS
        Frag16 pf;
        pf.u[0] = *(const v8u16*)&Pb[lo * LST + hi * 16];
        pf.u[1] = *(const v8u16*)&Pb[lo * LST + hi * 16 + 8];
        // ctx += P V (V fragments already resident)
        #pragma unroll
        for (int ni = 0; ni < 4; ni++)
            cacc[ni] = __builtin_amdgcn_wmma_f32_16x16x32_bf16(
                false, pf.bf, false, vf[ni].bf, (short)0, cacc[ni], false, false);
        __syncthreads();
    }
    // final 1/l normalization and store
    if (lane < 16) fac[lane] = 1.f / lrun;
    __syncthreads();
    #pragma unroll
    for (int r = 0; r < 8; r++) {
        float f = fac[r + 8 * hi];
        int gm = q0 + r + 8 * hi;
        size_t orow = (rowbase + gm) * DM + colbase;
        #pragma unroll
        for (int ni = 0; ni < 4; ni++)
            ctx[orow + ni * 16 + lo] = f2b(cacc[ni][r] * f);
    }
}

// ---------------------------------------------------------------- launch
extern "C" void kernel_launch(void* const* d_in, const int* in_sizes, int n_in,
                              void* d_out, int out_size, void* d_ws, size_t ws_size,
                              hipStream_t stream) {
    const float* x    = (const float*)d_in[0];
    const float* ln1s = (const float*)d_in[1];
    const float* ln1b = (const float*)d_in[2];
    const float* Wdq  = (const float*)d_in[3];
    const float* Wuq  = (const float*)d_in[4];
    const float* Wdkv = (const float*)d_in[5];
    const float* Wuk  = (const float*)d_in[6];
    const float* Wuv  = (const float*)d_in[7];
    const float* Wo   = (const float*)d_in[8];
    const float* bo   = (const float*)d_in[9];
    const float* ln2s = (const float*)d_in[10];
    const float* ln2b = (const float*)d_in[11];
    const float* W1   = (const float*)d_in[12];
    const float* b1   = (const float*)d_in[13];
    const float* W2   = (const float*)d_in[14];
    const float* b2   = (const float*)d_in[15];
    float* out = (float*)d_out;
    (void)in_sizes; (void)n_in; (void)out_size; (void)ws_size;

    const size_t M = 4096, C = 1024, DKV = 256, FF = 4096;
    char* ws = (char*)d_ws;
    size_t off = 0;
    auto take = [&](size_t bytes) -> char* {
        char* p = ws + off; off += (bytes + 255) & ~(size_t)255; return p;
    };
    u16*   h    = (u16*)take(M * C * 2);      // ln1 out; reused as h2
    u16*   cq   = (u16*)take(M * C * 2);      // latent q; reused as ctx
    u16*   qb   = (u16*)take(M * C * 2);
    u16*   ckv  = (u16*)take(M * DKV * 2);
    u16*   kb   = (u16*)take(M * C * 2);
    u16*   vtB  = (u16*)take(M * C * 2);      // V per-head transposed [B][H][64][SEQ]
    float* x1   = (float*)take(M * C * 4);
    u16*   ffb  = (u16*)take(M * FF * 2);
    u16*   wdqT = (u16*)take(C * C * 2);      // all weights stored transposed [N][K]
    u16*   wuqT = (u16*)take(C * C * 2);
    u16*   wdkT = (u16*)take(C * DKV * 2);
    u16*   wukT = (u16*)take(DKV * C * 2);
    u16*   wuvT = (u16*)take(DKV * C * 2);
    u16*   woT  = (u16*)take(C * C * 2);
    u16*   w1T  = (u16*)take(C * FF * 2);
    u16*   w2T  = (u16*)take(FF * C * 2);

    dim3 tb(32, 8);
    auto launch_cvt = [&](const float* src, u16* dst, size_t K_, size_t N_) {
        k_cvt_transpose<<<dim3((unsigned)(N_/32), (unsigned)(K_/32)), tb, 0, stream>>>(
            src, dst, (int)K_, (int)N_);
    };
    launch_cvt(Wdq,  wdqT, C,   C);
    launch_cvt(Wuq,  wuqT, C,   C);
    launch_cvt(Wdkv, wdkT, C,   DKV);
    launch_cvt(Wuk,  wukT, DKV, C);
    launch_cvt(Wuv,  wuvT, DKV, C);
    launch_cvt(Wo,   woT,  C,   C);
    launch_cvt(W1,   w1T,  C,   FF);
    launch_cvt(W2,   w2T,  FF,  C);

    dim3 blk(256);
    k_layernorm_bf16<<<dim3((unsigned)M), blk, 0, stream>>>(x, ln1s, ln1b, h, (int)C);

    k_gemm_bf16<0><<<dim3((unsigned)(M/128), (unsigned)(C/128)),   blk, 0, stream>>>(h,   wdqT, (int)M, (int)C,   (int)C,   nullptr, nullptr, cq,  nullptr);
    k_gemm_bf16<0><<<dim3((unsigned)(M/128), (unsigned)(C/128)),   blk, 0, stream>>>(cq,  wuqT, (int)M, (int)C,   (int)C,   nullptr, nullptr, qb,  nullptr);
    k_gemm_bf16<0><<<dim3((unsigned)(M/128), (unsigned)(DKV/128)), blk, 0, stream>>>(h,   wdkT, (int)M, (int)DKV, (int)C,   nullptr, nullptr, ckv, nullptr);
    k_gemm_bf16<0><<<dim3((unsigned)(M/128), (unsigned)(C/128)),   blk, 0, stream>>>(ckv, wukT, (int)M, (int)C,   (int)DKV, nullptr, nullptr, kb,  nullptr);
    k_gemm_bf16<3><<<dim3((unsigned)(M/128), (unsigned)(C/128)),   blk, 0, stream>>>(ckv, wuvT, (int)M, (int)C,   (int)DKV, nullptr, nullptr, vtB, nullptr);

    u16* ctx = cq;   // cq dead after q projection
    k_attention<<<dim3(4096), dim3(32), 0, stream>>>(qb, kb, vtB, ctx);

    k_gemm_bf16<1><<<dim3((unsigned)(M/128), (unsigned)(C/128)),   blk, 0, stream>>>(ctx, woT, (int)M, (int)C,  (int)C,  bo, x, nullptr, x1);

    u16* h2 = h;     // h dead after ckv
    k_layernorm_bf16<<<dim3((unsigned)M), blk, 0, stream>>>(x1, ln2s, ln2b, h2, (int)C);
    k_gemm_bf16<2><<<dim3((unsigned)(M/128), (unsigned)(FF/128)),  blk, 0, stream>>>(h2,  w1T, (int)M, (int)FF, (int)C,  b1, nullptr, ffb, nullptr);
    k_gemm_bf16<1><<<dim3((unsigned)(M/128), (unsigned)(C/128)),   blk, 0, stream>>>(ffb, w2T, (int)M, (int)C,  (int)FF, b2, x1, nullptr, out);
}
